// Reformer_26139170963885
// MI455X (gfx1250) — compile-verified, matrix operand order checked by reading, not run on hardware
//
#include <hip/hip_runtime.h>
#include <hip/hip_bf16.h>
#include <math.h>

// ---------------------------------------------------------------------------
// CDNA5 WMMA (wave32): V_WMMA_F32_16X16X32_BF16
// A: 16x32 bf16 (16 bf16/lane), B: 32x16 bf16, C/D: 16x16 f32 (8 f32/lane)
// Lane L of A needs row M=L%16, K elems {kh..kh+7} and {kh+16..kh+23},
// kh=(L/16)*8  ->  two contiguous 16B runs when tiles are stored K-major.
// ---------------------------------------------------------------------------
typedef __attribute__((ext_vector_type(16))) __bf16 v16bf;
typedef __attribute__((ext_vector_type(8)))  __bf16 v8bf;
typedef __attribute__((ext_vector_type(8)))  float  v8f;

static constexpr int T_LEN  = 4096;
static constexpr int EMB    = 512;
static constexpr int HEADS  = 8;
static constexpr int DHEAD  = 64;
static constexpr int BUCKET = 64;
static constexpr int NHASH  = 4;
static constexpr int BB     = 2;
static constexpr int BH     = BB * HEADS;       // 16
static constexpr int NSORT  = NHASH * T_LEN;    // 16384
static constexpr int CHUNKS = NSORT / BUCKET;   // 256

// Fragment loader from a K-major LDS tile (row stride `ld` bf16, ld % 8 == 0,
// so every (row*ld + k0 + kh) chunk is 16-byte aligned -> ds_load_b128).
// Used for A (row = M) and for B stored transposed [N][K] (row = N).
__device__ inline v16bf load_frag(const __bf16* s, int ld, int row0, int k0, int lane) {
    int m  = lane & 15;
    int kh = (lane >> 4) << 3;
    const __bf16* p = s + (size_t)(row0 + m) * ld + k0 + kh;
    v8bf lo = *(const v8bf*)(p);
    v8bf hi = *(const v8bf*)(p + 16);
    return __builtin_shufflevector(lo, hi, 0, 1, 2, 3, 4, 5, 6, 7,
                                   8, 9, 10, 11, 12, 13, 14, 15);
}

__device__ inline v8bf cvt8(const float* f) {
    v8bf r;
#pragma unroll
    for (int i = 0; i < 8; ++i) r[i] = (__bf16)f[i];
    return r;
}

// ---------------------------------------------------------------------------
// Weight pre-pass: W[K][N] f32 -> Wt[N][K] bf16 (LDS-tiled transpose+convert)
// grid = (N/32, K/32), block = 256
// ---------------------------------------------------------------------------
__global__ __launch_bounds__(256) void convert_w_k(
    const float* __restrict__ W, __bf16* __restrict__ Wt, int K, int N)
{
    __shared__ float tile[32][33];
    const int kb = blockIdx.y * 32, nb = blockIdx.x * 32;
    const int tx = threadIdx.x & 31, ty = threadIdx.x >> 5;   // 8 row-groups
    for (int r = ty; r < 32; r += 8)
        tile[r][tx] = W[(size_t)(kb + r) * N + nb + tx];
    __syncthreads();
    for (int r = ty; r < 32; r += 8)
        Wt[(size_t)(nb + r) * K + kb + tx] = (__bf16)tile[tx][r];
}

// ---------------------------------------------------------------------------
// GEMM: out = act(A[MxK] @ W (+bias) (+resid)); W given as Wt[N][K] bf16.
// Block = 128 threads (4 waves), 64x64 tile, K-step 32.
// flags: 1=bias, 2=relu, 4=residual add
// ---------------------------------------------------------------------------
#define TLD 40   // K-major tile row stride (bf16); 80 B = 5*16 -> 16B aligned

__global__ __launch_bounds__(128) void gemm_wmma_k(
    const float* __restrict__ A, const __bf16* __restrict__ Wt,
    const float* __restrict__ bias, const float* __restrict__ resid,
    float* __restrict__ out, int M, int N, int K, int flags)
{
    __shared__ __align__(16) __bf16 As[64 * TLD];
    __shared__ __align__(16) __bf16 Bs[64 * TLD];
    const int tid = threadIdx.x, lane = tid & 31, wave = tid >> 5;
    const int bm = blockIdx.y * 64, bn = blockIdx.x * 64;

    const int srow = tid >> 1;        // 0..63
    const int shalf = (tid & 1) * 16; // 0 or 16 (K offset)

    v8f zero8 = {0.f, 0.f, 0.f, 0.f, 0.f, 0.f, 0.f, 0.f};
    v8f acc[4] = {zero8, zero8, zero8, zero8};

    for (int k0 = 0; k0 < K; k0 += 32) {
        // stage A: 64x32 f32 -> bf16, coalesced float4 reads, v8bf stores
        {
            const float* ap = A + (size_t)(bm + srow) * K + k0 + shalf;
            float f[16];
#pragma unroll
            for (int q = 0; q < 4; ++q) {
                float4 t = *(const float4*)(ap + 4 * q);
                f[4 * q + 0] = t.x; f[4 * q + 1] = t.y;
                f[4 * q + 2] = t.z; f[4 * q + 3] = t.w;
            }
            *(v8bf*)&As[srow * TLD + shalf]     = cvt8(f);
            *(v8bf*)&As[srow * TLD + shalf + 8] = cvt8(f + 8);
        }
        // stage B: Wt[N][K] bf16, 128-bit loads and stores
        {
            const __bf16* wp = Wt + (size_t)(bn + srow) * K + k0 + shalf;
            v8bf b0 = *(const v8bf*)(wp);
            v8bf b1 = *(const v8bf*)(wp + 8);
            *(v8bf*)&Bs[srow * TLD + shalf]     = b0;
            *(v8bf*)&Bs[srow * TLD + shalf + 8] = b1;
        }
        if (k0 + 32 < K) {
            __builtin_prefetch(A + (size_t)(bm + srow) * K + k0 + 32, 0, 1);
            __builtin_prefetch(Wt + (size_t)(bn + srow) * K + k0 + 32, 0, 1);
        }
        __syncthreads();

        v16bf a = load_frag(As, TLD, wave * 16, 0, lane);
#pragma unroll
        for (int j = 0; j < 4; ++j) {
            v16bf b = load_frag(Bs, TLD, j * 16, 0, lane);
            acc[j] = __builtin_amdgcn_wmma_f32_16x16x32_bf16(
                false, a, false, b, (short)0, acc[j], false, false);
        }
        __syncthreads();
    }

    const int rbase = bm + wave * 16 + ((lane >> 4) << 3);
    const int cb    = bn + (lane & 15);
#pragma unroll
    for (int j = 0; j < 4; ++j) {
        int col = cb + j * 16;
        float bv = (flags & 1) ? bias[col] : 0.f;
#pragma unroll
        for (int r = 0; r < 8; ++r) {
            int row = rbase + r;
            float vv = acc[j][r] + bv;
            if (flags & 2) vv = fmaxf(vv, 0.f);
            if (flags & 4) vv += resid[(size_t)row * N + col];
            out[(size_t)row * N + col] = vv;
        }
    }
}

// ---------------------------------------------------------------------------
// LayerNorm (keras eps=1e-3), one 128-thread block per row of 512
// ---------------------------------------------------------------------------
__global__ __launch_bounds__(128) void layernorm_k(
    const float* __restrict__ x, const float* __restrict__ g,
    const float* __restrict__ b, float* __restrict__ out, int E)
{
    __shared__ float r1[128], r2[128];
    const int row = blockIdx.x, tid = threadIdx.x;
    float s = 0.f, s2 = 0.f;
    for (int i = tid; i < E; i += 128) {
        float v = x[(size_t)row * E + i];
        s += v; s2 += v * v;
    }
    r1[tid] = s; r2[tid] = s2;
    __syncthreads();
    for (int o = 64; o > 0; o >>= 1) {
        if (tid < o) { r1[tid] += r1[tid + o]; r2[tid] += r2[tid + o]; }
        __syncthreads();
    }
    float mean = r1[0] / E;
    float var  = r2[0] / E - mean * mean;
    float inv  = rsqrtf(var + 1e-3f);
    for (int i = tid; i < E; i += 128) {
        float v = x[(size_t)row * E + i];
        out[(size_t)row * E + i] = g[i] * (v - mean) * inv + b[i];
    }
}

// ---------------------------------------------------------------------------
// Counter-based normal generator for LSH hyperplanes R (b,d,NHASH,r)
// ---------------------------------------------------------------------------
__device__ inline unsigned int mix_u32(unsigned long long x) {
    x += 0x9E3779B97F4A7C15ULL;
    x = (x ^ (x >> 30)) * 0xBF58476D1CE4E5B9ULL;
    x = (x ^ (x >> 27)) * 0x94D049BB133111EBULL;
    return (unsigned int)(x ^ (x >> 31));
}

__global__ void gen_normal_k(float* __restrict__ R, int n, unsigned int seed)
{
    int i = blockIdx.x * blockDim.x + threadIdx.x;
    if (i >= n) return;
    unsigned long long base = ((unsigned long long)seed << 33);
    unsigned int u1 = mix_u32(base | (unsigned long long)(2 * i));
    unsigned int u2 = mix_u32(base | (unsigned long long)(2 * i + 1));
    float f1 = ((float)u1 + 1.0f) * (1.0f / 4294967296.0f);
    float f2 = (float)u2 * (1.0f / 4294967296.0f);
    R[i] = sqrtf(-2.0f * __logf(f1)) * __cosf(6.28318530718f * f2);
}

// ---------------------------------------------------------------------------
// LSH bucket keys, exact f32 (argmax is bit-sensitive). grid = (NHASH, BH)
// key[bh][hash*T + t] = bucket * T + t
// ---------------------------------------------------------------------------
__global__ __launch_bounds__(256) void bucket_keys_k(
    const float* __restrict__ qk, const float* __restrict__ R,
    int* __restrict__ keys)
{
    const int hash = blockIdx.x, bh = blockIdx.y;
    const int b = bh / HEADS, h = bh % HEADS;
    __shared__ float Rs[DHEAD * 32];
    for (int i = threadIdx.x; i < DHEAD * 32; i += 256) {
        int f = i >> 5, r = i & 31;
        Rs[i] = R[(((size_t)bh * DHEAD + f) * NHASH + hash) * 32 + r];
    }
    __syncthreads();
    for (int t = threadIdx.x; t < T_LEN; t += 256) {
        const float* q = qk + ((size_t)b * T_LEN + t) * EMB + h * DHEAD;
        float s[32];
#pragma unroll
        for (int r = 0; r < 32; ++r) s[r] = 0.f;
        for (int f = 0; f < DHEAD; ++f) {
            float qv = q[f];
#pragma unroll
            for (int r = 0; r < 32; ++r) s[r] += qv * Rs[f * 32 + r];
        }
        float best = s[0]; int bi = 0;
#pragma unroll
        for (int r = 1; r < 32; ++r) if (s[r] > best) { best = s[r]; bi = r; }
#pragma unroll
        for (int r = 0; r < 32; ++r) if (-s[r] > best) { best = -s[r]; bi = 32 + r; }
        int bucket = bi + 64 * hash;
        keys[(size_t)bh * NSORT + hash * T_LEN + t] = bucket * T_LEN + t;
    }
}

// ---------------------------------------------------------------------------
// Bitonic sort of 16384 (key, idx) pairs per head-batch in 128 KB dynamic LDS
// (CDNA5 WGP: 320 KB). grid = BH, block = 1024.
// ---------------------------------------------------------------------------
__global__ __launch_bounds__(1024) void sort_k(
    const int* __restrict__ keys, int* __restrict__ sortedJ, int n)
{
    extern __shared__ char dsm[];
    int* skey = (int*)dsm;
    int* sidx = skey + n;
    const int bh = blockIdx.x;
    const int* kk = keys + (size_t)bh * n;
    for (int i = threadIdx.x; i < n; i += 1024) { skey[i] = kk[i]; sidx[i] = i; }
    __syncthreads();
    for (int k = 2; k <= n; k <<= 1) {
        for (int j = k >> 1; j > 0; j >>= 1) {
            for (int i = threadIdx.x; i < n; i += 1024) {
                int l = i ^ j;
                if (l > i) {
                    bool up = ((i & k) == 0);
                    int ki = skey[i], kl = skey[l];
                    if (up ? (ki > kl) : (ki < kl)) {
                        skey[i] = kl; skey[l] = ki;
                        int t0 = sidx[i]; sidx[i] = sidx[l]; sidx[l] = t0;
                    }
                }
            }
            __syncthreads();
        }
    }
    for (int i = threadIdx.x; i < n; i += 1024)
        sortedJ[(size_t)bh * n + i] = sidx[i];
}

// ---------------------------------------------------------------------------
// Chunked LSH attention. grid = (CHUNKS, BH), block = 128 (4 waves).
// All WMMA operands staged K-major in LDS so fragments are ds_load_b128 pairs:
//   sq[q][f] (A of QK^T), sk[kv][f] (B of QK^T, rows normalized),
//   probs[q][kv] (A of P@V), svT[f][kv] (B of P@V).
// ---------------------------------------------------------------------------
static constexpr int LDF = 72;    // 64-wide K-major rows (f dim), 144 B
static constexpr int LDK = 136;   // 128-wide K-major rows (kv dim), 272 B

__global__ __launch_bounds__(128) void lsh_attn_k(
    const float* __restrict__ qk, const float* __restrict__ v,
    const int* __restrict__ sortedJ,
    float* __restrict__ o_heads, float* __restrict__ lsebuf)
{
    extern __shared__ char dsm[];
    int*    jfull = (int*)dsm;                        // 64
    int*    qpos  = jfull + 64;                       // 64
    int*    kpos  = qpos + 64;                        // 128
    __bf16* sq    = (__bf16*)(kpos + 128);            // 64  * LDF
    __bf16* sk    = sq + 64 * LDF;                    // 128 * LDF
    __bf16* svT   = sk + 128 * LDF;                   // 64  * LDK
    __bf16* probs = svT + 64 * LDK;                   // 64  * LDK
    float*  dots  = (float*)(probs + 64 * LDK);       // 64 * 128

    const int chunk = blockIdx.x, bh = blockIdx.y;
    const int b = bh / HEADS, h = bh % HEADS;
    const int tid = threadIdx.x, lane = tid & 31, wave = tid >> 5;
    const int* sj = sortedJ + (size_t)bh * NSORT;
    const int prev = (chunk + CHUNKS - 1) % CHUNKS;

    if (tid < 64) {
        int jf = sj[chunk * BUCKET + tid];
        jfull[tid] = jf;
        qpos[tid] = jf % T_LEN;
    }
    if (tid < 128) {
        int src = (tid < 64) ? chunk : prev;
        kpos[tid] = sj[src * BUCKET + (tid & 63)] % T_LEN;
    }
    __syncthreads();

    // stage sq: two threads per q row, 32 f each, float4 reads -> v8bf stores
    {
        int qi = tid >> 1, half = (tid & 1) * 32;
        const float* qr = qk + ((size_t)b * T_LEN + qpos[qi]) * EMB + h * DHEAD + half;
        float f[32];
#pragma unroll
        for (int q = 0; q < 8; ++q) {
            float4 t = *(const float4*)(qr + 4 * q);
            f[4 * q + 0] = t.x; f[4 * q + 1] = t.y;
            f[4 * q + 2] = t.z; f[4 * q + 3] = t.w;
        }
#pragma unroll
        for (int q = 0; q < 4; ++q)
            *(v8bf*)&sq[qi * LDF + half + 8 * q] = cvt8(f + 8 * q);
    }
    // stage sk (normalized rows, K-major) and svT (transposed scalar stores)
    {
        int kv = tid;   // 128 threads <-> 128 kv rows
        const float* kr = qk + ((size_t)b * T_LEN + kpos[kv]) * EMB + h * DHEAD;
        const float* vr = v  + ((size_t)b * T_LEN + kpos[kv]) * EMB + h * DHEAD;
        float tmp[DHEAD];
        float ss = 0.f;
#pragma unroll
        for (int f = 0; f < DHEAD; ++f) { tmp[f] = kr[f]; ss += tmp[f] * tmp[f]; }
        float rn = rsqrtf(ss);
#pragma unroll
        for (int f = 0; f < DHEAD; ++f) tmp[f] *= rn;
#pragma unroll
        for (int q = 0; q < 8; ++q)
            *(v8bf*)&sk[kv * LDF + 8 * q] = cvt8(tmp + 8 * q);
#pragma unroll
        for (int f = 0; f < DHEAD; ++f)
            svT[f * LDK + kv] = (__bf16)vr[f];
    }
    __syncthreads();

    // dots = sq @ sk^T * d^-0.5, self-mask -> LDS f32
    v8f zero8 = {0.f, 0.f, 0.f, 0.f, 0.f, 0.f, 0.f, 0.f};
    for (int mt = 0; mt < 4; ++mt) {
#pragma unroll
        for (int nt = 0; nt < 2; ++nt) {
            v8f acc = zero8;
#pragma unroll
            for (int ks = 0; ks < 2; ++ks) {
                v16bf a  = load_frag(sq, LDF, mt * 16, ks * 32, lane);
                v16bf bb = load_frag(sk, LDF, wave * 32 + nt * 16, ks * 32, lane);
                acc = __builtin_amdgcn_wmma_f32_16x16x32_bf16(
                    false, a, false, bb, (short)0, acc, false, false);
            }
            int row0 = mt * 16 + ((lane >> 4) << 3);
            int col  = wave * 32 + nt * 16 + (lane & 15);
#pragma unroll
            for (int r = 0; r < 8; ++r) {
                int row = row0 + r;
                float dv = acc[r] * 0.125f;              // d^-0.5, d = 64
                if (qpos[row] == kpos[col]) dv = -1e5f;  // self mask
                dots[row * 128 + col] = dv;
            }
        }
    }
    __syncthreads();

    // row softmax + LSE, scatter LSE via sortedJ (implicit undo)
    if (tid < 64) {
        float m = -3.4e38f;
        for (int j = 0; j < 128; ++j) m = fmaxf(m, dots[tid * 128 + j]);
        float sum = 0.f;
        for (int j = 0; j < 128; ++j) sum += __expf(dots[tid * 128 + j] - m);
        float lse = m + __logf(sum);
        for (int j = 0; j < 128; ++j)
            probs[tid * LDK + j] = (__bf16)__expf(dots[tid * 128 + j] - lse);
        lsebuf[(size_t)bh * NSORT + jfull[tid]] = lse;
    }
    __syncthreads();

    // o = probs @ svf, scatter rows via sortedJ
    for (int mt = 0; mt < 4; ++mt) {
        v8f acc = zero8;
#pragma unroll
        for (int ks = 0; ks < 4; ++ks) {
            v16bf a  = load_frag(probs, LDK, mt * 16, ks * 32, lane);
            v16bf bb = load_frag(svT, LDK, wave * 16, ks * 32, lane);
            acc = __builtin_amdgcn_wmma_f32_16x16x32_bf16(
                false, a, false, bb, (short)0, acc, false, false);
        }
        int row0 = mt * 16 + ((lane >> 4) << 3);
        int col  = wave * 16 + (lane & 15);
#pragma unroll
        for (int r = 0; r < 8; ++r) {
            int row = row0 + r;
            o_heads[((size_t)bh * NSORT + jfull[row]) * DHEAD + col] = acc[r];
        }
    }
}

// ---------------------------------------------------------------------------
// Combine NHASH rounds with softmax(lse) weights. grid = (T/256, BH)
// ---------------------------------------------------------------------------
__global__ __launch_bounds__(256) void combine_k(
    const float* __restrict__ o_heads, const float* __restrict__ lsebuf,
    float* __restrict__ attnC)
{
    const int t = blockIdx.x * 256 + threadIdx.x;
    const int bh = blockIdx.y;
    if (t >= T_LEN) return;
    float l[NHASH];
    float m = -3.4e38f;
#pragma unroll
    for (int hh = 0; hh < NHASH; ++hh) {
        l[hh] = lsebuf[(size_t)bh * NSORT + hh * T_LEN + t];
        m = fmaxf(m, l[hh]);
    }
    float s = 0.f;
#pragma unroll
    for (int hh = 0; hh < NHASH; ++hh) { l[hh] = __expf(l[hh] - m); s += l[hh]; }
    float inv = 1.f / s;
    for (int dd = 0; dd < DHEAD; ++dd) {
        float acc = 0.f;
#pragma unroll
        for (int hh = 0; hh < NHASH; ++hh)
            acc += o_heads[((size_t)bh * NSORT + hh * T_LEN + t) * DHEAD + dd] * (l[hh] * inv);
        attnC[((size_t)bh * T_LEN + t) * DHEAD + dd] = acc;
    }
}

// ---------------------------------------------------------------------------
// TF split_heads reinterpretation shuffle
// ---------------------------------------------------------------------------
__global__ void shuffle_k(const float* __restrict__ attnC, float* __restrict__ z)
{
    size_t i = (size_t)blockIdx.x * 256 + threadIdx.x;
    const size_t per = (size_t)T_LEN * EMB;
    if (i >= (size_t)BB * per) return;
    int b = (int)(i / per);
    size_t o = i % per;
    int h2 = (int)(o / ((size_t)T_LEN * DHEAD));
    int t2 = (int)((o / DHEAD) % T_LEN);
    int d2 = (int)(o % DHEAD);
    z[i] = attnC[(size_t)b * per + (size_t)t2 * EMB + h2 * DHEAD + d2];
}

__global__ void add_k(const float* __restrict__ a, const float* __restrict__ b,
                      float* __restrict__ out, size_t n)
{
    size_t i = (size_t)blockIdx.x * 256 + threadIdx.x;
    if (i < n) out[i] = a[i] + b[i];
}

// ---------------------------------------------------------------------------
// Host orchestration
// ---------------------------------------------------------------------------
extern "C" void kernel_launch(void* const* d_in, const int* in_sizes, int n_in,
                              void* d_out, int out_size, void* d_ws, size_t ws_size,
                              hipStream_t stream)
{
    (void)in_sizes; (void)n_in; (void)out_size; (void)ws_size;
    const float* x = (const float*)d_in[0];
    const int M = BB * T_LEN;                 // 8192 rows
    const size_t actB = (size_t)M * EMB * sizeof(float);       // 16.78 MB
    const size_t bigB = (size_t)M * 4 * EMB * sizeof(float);   // 67.1 MB

    char* ws = (char*)d_ws;
    size_t off = 0;
    auto carve = [&](size_t bytes) { char* p = ws + off; off += (bytes + 255) & ~(size_t)255; return p; };

    float*  y1    = (float*)carve(actB);
    float*  y2    = (float*)carve(actB);
    float*  qkb   = (float*)carve(actB);   // qk projection; later FFN f2
    float*  vb    = (float*)carve(actB);   // v projection
    float*  zb    = (float*)carve(actB);   // shuffled attn; later LN out
    float*  attnC = (float*)carve(actB);
    float*  big   = (float*)carve(bigB);   // o_heads, then FFN hidden f1/f3
    __bf16* wkT   = (__bf16*)carve((size_t)EMB * EMB * sizeof(__bf16));
    __bf16* wvT   = (__bf16*)carve((size_t)EMB * EMB * sizeof(__bf16));
    __bf16* woT   = (__bf16*)carve((size_t)EMB * EMB * sizeof(__bf16));
    __bf16* w1T   = (__bf16*)carve((size_t)EMB * 4 * EMB * sizeof(__bf16));
    __bf16* w2T   = (__bf16*)carve((size_t)EMB * 4 * EMB * sizeof(__bf16));
    float*  Rbuf  = (float*)carve((size_t)BH * DHEAD * NHASH * 32 * sizeof(float));
    int*    keys  = (int*)carve((size_t)BH * NSORT * sizeof(int));
    int*    sortJ = (int*)carve((size_t)BH * NSORT * sizeof(int));
    float*  lseb  = (float*)carve((size_t)BH * NSORT * sizeof(float));

    const size_t attnSmem = (64 + 64 + 128) * sizeof(int)
                          + (64 * LDF + 128 * LDF) * sizeof(__bf16)
                          + (64 * LDK + 64 * LDK) * sizeof(__bf16)
                          + 64 * 128 * sizeof(float);          // ~95 KB
    const size_t sortSmem = 2 * (size_t)NSORT * sizeof(int);   // 128 KB

    const float* x1s = x;
    const float* x2s = x;

    for (int layer = 0; layer < 2; ++layer) {
        const int base = 1 + layer * 10;
        const float* Wk  = (const float*)d_in[base + 0];
        const float* Wv  = (const float*)d_in[base + 1];
        const float* Wo  = (const float*)d_in[base + 2];
        const float* bo  = (const float*)d_in[base + 3];
        const float* lng = (const float*)d_in[base + 4];
        const float* lnb = (const float*)d_in[base + 5];
        const float* W1  = (const float*)d_in[base + 6];
        const float* b1  = (const float*)d_in[base + 7];
        const float* W2  = (const float*)d_in[base + 8];
        const float* b2  = (const float*)d_in[base + 9];

        // --- weight pre-pass: transpose + bf16 convert (reused by 128 blocks) ---
        convert_w_k<<<dim3(EMB / 32, EMB / 32), 256, 0, stream>>>(Wk, wkT, EMB, EMB);
        convert_w_k<<<dim3(EMB / 32, EMB / 32), 256, 0, stream>>>(Wv, wvT, EMB, EMB);
        convert_w_k<<<dim3(EMB / 32, EMB / 32), 256, 0, stream>>>(Wo, woT, EMB, EMB);
        convert_w_k<<<dim3(4 * EMB / 32, EMB / 32), 256, 0, stream>>>(W1, w1T, EMB, 4 * EMB);
        convert_w_k<<<dim3(EMB / 32, 4 * EMB / 32), 256, 0, stream>>>(W2, w2T, 4 * EMB, EMB);

        // --- LSH hyperplanes ---
        {
            int n = BH * DHEAD * NHASH * 32;
            gen_normal_k<<<(n + 255) / 256, 256, 0, stream>>>(Rbuf, n, 42u + (unsigned)layer);
        }
        // --- projections (bf16 WMMA) ---
        gemm_wmma_k<<<dim3(EMB / 64, M / 64), 128, 0, stream>>>(
            x2s, wkT, nullptr, nullptr, qkb, M, EMB, EMB, 0);
        gemm_wmma_k<<<dim3(EMB / 64, M / 64), 128, 0, stream>>>(
            x2s, wvT, nullptr, nullptr, vb, M, EMB, EMB, 0);
        // --- buckets / sort keys ---
        bucket_keys_k<<<dim3(NHASH, BH), 256, 0, stream>>>(qkb, Rbuf, keys);
        sort_k<<<BH, 1024, sortSmem, stream>>>(keys, sortJ, NSORT);
        // --- chunked attention (WMMA) ---
        lsh_attn_k<<<dim3(CHUNKS, BH), 128, attnSmem, stream>>>(
            qkb, vb, sortJ, big /* o_heads */, lseb);
        // --- multi-round combine + head shuffle ---
        combine_k<<<dim3(T_LEN / 256, BH), 256, 0, stream>>>(big, lseb, attnC);
        {
            size_t tot = (size_t)BB * T_LEN * EMB;
            shuffle_k<<<(unsigned)((tot + 255) / 256), 256, 0, stream>>>(attnC, zb);
        }
        // --- y1 = x1 + z @ Wo + bo ---
        gemm_wmma_k<<<dim3(EMB / 64, M / 64), 128, 0, stream>>>(
            zb, woT, bo, x1s, y1, M, EMB, EMB, 1 | 4);
        // --- g(y1) = ffn(ffn(layernorm(y1))) ; y2 = x2 + g(y1) ---
        layernorm_k<<<M, 128, 0, stream>>>(y1, lng, lnb, zb, EMB);
        gemm_wmma_k<<<dim3(4 * EMB / 64, M / 64), 128, 0, stream>>>(
            zb, w1T, b1, nullptr, big, M, 4 * EMB, EMB, 1 | 2);
        gemm_wmma_k<<<dim3(EMB / 64, M / 64), 128, 0, stream>>>(
            big, w2T, b2, nullptr, qkb, M, EMB, 4 * EMB, 1);
        gemm_wmma_k<<<dim3(4 * EMB / 64, M / 64), 128, 0, stream>>>(
            qkb, w1T, b1, nullptr, big, M, 4 * EMB, EMB, 1 | 2);
        gemm_wmma_k<<<dim3(EMB / 64, M / 64), 128, 0, stream>>>(
            big, w2T, b2, x2s, y2, M, EMB, 4 * EMB, 1 | 4);

        x1s = y1;
        x2s = y2;
    }

    {
        size_t n = (size_t)BB * T_LEN * EMB;
        add_k<<<(unsigned)((n + 255) / 256), 256, 0, stream>>>(y1, y2, (float*)d_out, n);
    }
}
